// LSTMGenerator_10136122818873
// MI455X (gfx1250) — matrix operand order repaired and checked by reading, not compiled, not run on hardware
//
#include <hip/hip_runtime.h>

// ext-vector types matching CDNA5 WMMA operand shapes (wave32)
typedef __attribute__((ext_vector_type(16))) _Float16 v16h;
typedef __attribute__((ext_vector_type(8)))  _Float16 v8h;
typedef __attribute__((ext_vector_type(4)))  _Float16 v4h;
typedef __attribute__((ext_vector_type(4)))  float    v4f;
typedef __attribute__((ext_vector_type(8)))  float    v8f;

#define LOG2E 1.44269504088896340736f

// fast sigmoid/tanh on v_exp_f32 + v_rcp_f32 (no IEEE div sequence)
__device__ __forceinline__ float sigf(float x) {
    return __builtin_amdgcn_rcpf(1.0f + __builtin_amdgcn_exp2f(-LOG2E * x));
}
__device__ __forceinline__ float tanh_fast(float x) {
    return 1.0f - 2.0f * __builtin_amdgcn_rcpf(1.0f + __builtin_amdgcn_exp2f(2.0f * LOG2E * x));
}

__device__ __forceinline__ v16h cat8(v8h lo, v8h hi) {
    union { v16h v; v8h h[2]; } u;
    u.h[0] = lo; u.h[1] = hi;
    return u.v;
}

// raw 4-element global load type per input precision
template<typename T> struct Raw4;
template<> struct Raw4<float>    { typedef v4f type; };
template<> struct Raw4<_Float16> { typedef v4h type; };

__device__ __forceinline__ v4h to4h(v4f a) {
    v4h r;
    #pragma unroll
    for (int e = 0; e < 4; ++e) r[e] = (_Float16)a[e];
    return r;
}
__device__ __forceinline__ v4h to4h(v4h a) { return a; }

__device__ __forceinline__ void store_out(float* p, float v)    { *p = v; }
__device__ __forceinline__ void store_out(_Float16* p, float v) { *p = (_Float16)v; }

// One LSTM layer, full sequential scan over T.
//   - grid.x = B/16 workgroups, each owns 16 batch rows for the whole sequence
//   - NW = H/16 waves; wave w owns hidden columns [16w,16w+16) and its 4 gate tiles
//   - c state in VGPRs (f32), h state in LDS (f16), Whh in LDS (f16), Wih frags in VGPRs
//   - x double-buffered in LDS; x_{t+1} prefetched through registers during step t
template<typename TIN, typename TOUT, int KIN, int H>
__global__ __launch_bounds__((H / 16) * 32)
void lstm_scan_kernel(const TIN* __restrict__ xin,    // [B,T,KIN]
                      const float* __restrict__ Wih,  // [4H,KIN]
                      const float* __restrict__ Whh,  // [4H,H]
                      const float* __restrict__ bih,  // [4H]
                      const float* __restrict__ bhh,  // [4H]
                      TOUT* __restrict__ hout,        // [B,T,H]
                      int T)
{
    constexpr int NW   = H / 16;          // waves per block
    constexpr int NTH  = NW * 32;         // threads per block
    constexpr int SA   = 2 * KIN + H + 8; // A row stride (halves): [x ping | x pong | h]
    constexpr int SW   = H + 8;           // Whh row stride (halves)
    constexpr int KSX  = KIN / 32;        // WMMA k-steps over x part
    constexpr int KSH  = H / 32;          // WMMA k-steps over h part
    constexpr int NCH  = 16 * KIN / 4;    // 4-element chunks per timestep tile
    constexpr int NPF  = NCH / NTH;       // chunks prefetched per thread
    static_assert(NCH % NTH == 0, "chunk count must divide thread count");

    __shared__ _Float16 Ash[16 * SA];     // [x0 | x1 | h] tile, 16 batch rows
    __shared__ _Float16 Wsh[4 * H * SW];  // Whh as f16, row = gate output index

    const int tid  = threadIdx.x;
    const int lane = tid & 31;
    const int wid  = tid >> 5;
    const int hi   = lane >> 4;
    const int col  = lane & 15;
    const int jb   = wid * 16 + col;      // hidden index owned by this lane
    const int b0   = blockIdx.x * 16;

    // this thread's prefetch chunk coordinates (row m, element offset k)
    int cm[NPF], ck[NPF];
    #pragma unroll
    for (int i = 0; i < NPF; ++i) {
        int cid = tid + i * NTH;
        cm[i] = cid / (KIN / 4);
        ck[i] = (cid % (KIN / 4)) * 4;
    }
    typename Raw4<TIN>::type pf[NPF];

    // ---- prologue ----------------------------------------------------------
    // issue x_0 loads first so their latency overlaps the weight packing
    #pragma unroll
    for (int i = 0; i < NPF; ++i)
        pf[i] = *(const typename Raw4<TIN>::type*)
                 (xin + ((size_t)(b0 + cm[i]) * T + 0) * KIN + ck[i]);

    // pack Whh (f32 -> f16) into LDS
    for (int idx = tid; idx < 4 * H * H; idx += NTH) {
        int g = idx / H, k = idx - g * H;
        Wsh[g * SW + k] = (_Float16)Whh[idx];
    }
    // zero the A buffer (h region must start at 0, matches h0 = 0)
    for (int idx = tid; idx < 16 * SA; idx += NTH) Ash[idx] = (_Float16)0.0f;

    // preload time-invariant Wih B-fragments into registers
    // B 32x16 layout: lane = column n, halves = K (ks*32 + hi*16 + 0..15), contiguous
    v16h wx[4][KSX];
    #pragma unroll
    for (int q = 0; q < 4; ++q) {
        #pragma unroll
        for (int ks = 0; ks < KSX; ++ks) {
            const float* wr = Wih + (size_t)(q * H + jb) * KIN + ks * 32 + hi * 16;
            v16h f;
            #pragma unroll
            for (int e = 0; e < 16; ++e) f[e] = (_Float16)wr[e];
            wx[q][ks] = f;
        }
    }

    // per-lane gate bias splatted into accumulator-init vectors (WMMA C operand)
    v8f acc0[4];
    #pragma unroll
    for (int q = 0; q < 4; ++q) {
        float b = bih[q * H + jb] + bhh[q * H + jb];
        #pragma unroll
        for (int r = 0; r < 8; ++r) acc0[q][r] = b;
    }

    // cell state; C/D tile layout: VGPR r, lane -> (row r + 8*hi, col)
    float creg[8];
    #pragma unroll
    for (int r = 0; r < 8; ++r) creg[r] = 0.0f;

    // stash x_0 into ping buffer
    #pragma unroll
    for (int i = 0; i < NPF; ++i)
        *(v4h*)&Ash[cm[i] * SA + ck[i]] = to4h(pf[i]);

    const _Float16* arow = &Ash[(lane & 15) * SA];  // A row owned by this lane

    __syncthreads();

    // ---- sequential scan ----------------------------------------------------
    for (int t = 0; t < T; ++t) {
        // prefetch x_{t+1}: latency overlaps the entire WMMA + gate-math body
        if (t + 1 < T) {
            #pragma unroll
            for (int i = 0; i < NPF; ++i)
                pf[i] = *(const typename Raw4<TIN>::type*)
                         (xin + ((size_t)(b0 + cm[i]) * T + (t + 1)) * KIN + ck[i]);
        }

        const int xo = (t & 1) * KIN;  // this step's x buffer

        v8f acc[4];
        // x-part, first k-step consumes the bias-init C operand directly
        {
            v8h alo = *(const v8h*)(arow + xo + hi * 16);
            v8h ahi = *(const v8h*)(arow + xo + hi * 16 + 8);
            v16h a = cat8(alo, ahi);
            #pragma unroll
            for (int q = 0; q < 4; ++q)
                acc[q] = __builtin_amdgcn_wmma_f32_16x16x32_f16(
                    false, a, false, wx[q][0], (short)0, acc0[q], false, false);
        }
        #pragma unroll
        for (int ks = 1; ks < KSX; ++ks) {
            v8h alo = *(const v8h*)(arow + xo + ks * 32 + hi * 16);
            v8h ahi = *(const v8h*)(arow + xo + ks * 32 + hi * 16 + 8);
            v16h a = cat8(alo, ahi);
            #pragma unroll
            for (int q = 0; q < 4; ++q)
                acc[q] = __builtin_amdgcn_wmma_f32_16x16x32_f16(
                    false, a, false, wx[q][ks], (short)0, acc[q], false, false);
        }
        // h-part: A from LDS, B (Whh) from LDS
        #pragma unroll
        for (int ks = 0; ks < KSH; ++ks) {
            v8h alo = *(const v8h*)(arow + 2 * KIN + ks * 32 + hi * 16);
            v8h ahi = *(const v8h*)(arow + 2 * KIN + ks * 32 + hi * 16 + 8);
            v16h a = cat8(alo, ahi);
            #pragma unroll
            for (int q = 0; q < 4; ++q) {
                const _Float16* brow = &Wsh[(size_t)(q * H + jb) * SW + ks * 32 + hi * 16];
                v16h b = cat8(*(const v8h*)brow, *(const v8h*)(brow + 8));
                acc[q] = __builtin_amdgcn_wmma_f32_16x16x32_f16(
                    false, a, false, b, (short)0, acc[q], false, false);
            }
        }
        __syncthreads();  // all waves done reading this step's LDS regions

        // gate math + state update; write h back to LDS (f16) and global
        #pragma unroll
        for (int r = 0; r < 8; ++r) {
            float cn = sigf(acc[1][r]) * creg[r] + sigf(acc[0][r]) * tanh_fast(acc[2][r]);
            creg[r] = cn;
            float hn = sigf(acc[3][r]) * tanh_fast(cn);
            int m = hi * 8 + r;
            Ash[m * SA + 2 * KIN + jb] = (_Float16)hn;
            store_out(hout + ((size_t)(b0 + m) * T + t) * H + jb, hn);
        }

        // stash prefetched x_{t+1} into the other x buffer (disjoint from readers)
        if (t + 1 < T) {
            const int xn = ((t + 1) & 1) * KIN;
            #pragma unroll
            for (int i = 0; i < NPF; ++i)
                *(v4h*)&Ash[cm[i] * SA + xn + ck[i]] = to4h(pf[i]);
        }
        __syncthreads();  // h + next-x writes visible before next step's reads
    }
}

// out[b,t] = tanh(rec[b,t,:] . Wlin + blin) -- memory-bound dot, one thread per (b,t)
__global__ void linear_tanh_kernel(const float* __restrict__ rec,   // [B*T,128]
                                   const float* __restrict__ Wlin,  // [1,128]
                                   const float* __restrict__ blin,  // [1]
                                   float* __restrict__ out, int N)
{
    int i = blockIdx.x * blockDim.x + threadIdx.x;
    if (i >= N) return;
    const float4* r4 = (const float4*)(rec + (size_t)i * 128);
    const float4* w4 = (const float4*)Wlin;
    float s = 0.0f;
    #pragma unroll
    for (int k = 0; k < 32; ++k) {
        float4 a = r4[k], b = w4[k];
        s += a.x * b.x + a.y * b.y + a.z * b.z + a.w * b.w;
    }
    out[i] = tanh_fast(s + blin[0]);
}

extern "C" void kernel_launch(void* const* d_in, const int* in_sizes, int n_in,
                              void* d_out, int out_size, void* d_ws, size_t ws_size,
                              hipStream_t stream) {
    (void)in_sizes; (void)n_in; (void)out_size; (void)ws_size;
    const int B = 256, T = 1024;

    const float* x    = (const float*)d_in[0];
    const float* Wih0 = (const float*)d_in[1];
    const float* Whh0 = (const float*)d_in[2];
    const float* bih0 = (const float*)d_in[3];
    const float* bhh0 = (const float*)d_in[4];
    const float* Wih1 = (const float*)d_in[5];
    const float* Whh1 = (const float*)d_in[6];
    const float* bih1 = (const float*)d_in[7];
    const float* bhh1 = (const float*)d_in[8];
    const float* Wih2 = (const float*)d_in[9];
    const float* Whh2 = (const float*)d_in[10];
    const float* bih2 = (const float*)d_in[11];
    const float* bhh2 = (const float*)d_in[12];
    const float* Wlin = (const float*)d_in[13];
    const float* blin = (const float*)d_in[14];

    float* out = (float*)d_out;                    // [B,T,1]
    float* rec = out + (size_t)B * T;              // [B,T,128] (second return value)

    // inter-layer activations kept in f16: halves traffic, no per-step converts
    _Float16* h0 = (_Float16*)d_ws;                // [B,T,32]  f16 = 16 MB
    _Float16* h1 = h0 + (size_t)B * T * 32;        // [B,T,64]  f16 = 32 MB

    lstm_scan_kernel<float,    _Float16, 32, 32 ><<<B / 16,  64, 0, stream>>>(x,  Wih0, Whh0, bih0, bhh0, h0,  T);
    lstm_scan_kernel<_Float16, _Float16, 32, 64 ><<<B / 16, 128, 0, stream>>>(h0, Wih1, Whh1, bih1, bhh1, h1,  T);
    lstm_scan_kernel<_Float16, float,    64, 128><<<B / 16, 256, 0, stream>>>(h1, Wih2, Whh2, bih2, bhh2, rec, T);
    linear_tanh_kernel<<<(B * T + 255) / 256, 256, 0, stream>>>(rec, Wlin, blin, out, B * T);
}
